// ConvLSTMCell_12558484374243
// MI455X (gfx1250) — matrix operand order repaired and checked by reading, not compiled, not run on hardware
//
#include <hip/hip_runtime.h>
#include <hip/hip_bf16.h>

typedef __attribute__((ext_vector_type(16))) __bf16       v16bf;
typedef __attribute__((ext_vector_type(8)))  float        v8f;
typedef __attribute__((ext_vector_type(4)))  unsigned int uint4v;

#define CIN    21
#define HCC    32
#define CTOT   53          // 21 + 32 real channels
#define CH4    128         // 4*Hc output channels (M)
#define IMG    256
#define PLANE  65536       // IMG*IMG
#define NTAP   9
#define CSLOT  64          // channel slot per tap in packed K (53 real + 11 pad)
#define KTOT   576         // 9 taps * 64 = 18 chunks of 32
#define KCH    18
#define TILE_N 64
#define TROWS  3
#define TCOLS  66          // 64 + halo
#define TCH    72          // padded channel stride (bf16) -> 144B col stride, bank-conflict-free
#define GPAD   68          // g-tile row stride (floats)

__device__ __forceinline__ unsigned short f2bf(float f) {
    unsigned int u = __builtin_bit_cast(unsigned int, f);
    u = (u + 0x7FFFu + ((u >> 16) & 1u)) >> 16;   // round-to-nearest-even
    return (unsigned short)u;
}

// ---- pack wx[128,21,3,3] & wh[128,32,3,3] -> bf16 Wp[128][576], k = tap*64 + ch ----
__global__ void pack_weights_kernel(const float* __restrict__ wx,
                                    const float* __restrict__ wh,
                                    unsigned short* __restrict__ Wp) {
    int i = blockIdx.x * blockDim.x + threadIdx.x;
    if (i >= CH4 * KTOT) return;
    int m   = i / KTOT;
    int k   = i - m * KTOT;
    int tap = k >> 6;          // 0..8  (= r*3+s)
    int ch  = k & 63;
    float v = 0.f;
    if (ch < CIN)       v = wx[m * (CIN * 9) + ch * 9 + tap];
    else if (ch < CTOT) v = wh[m * (HCC * 9) + (ch - CIN) * 9 + tap];
    Wp[i] = f2bf(v);
}

// ---- fused implicit-GEMM ConvLSTM cell ----
__launch_bounds__(256)
__global__ void convlstm_wmma_kernel(const float* __restrict__ xin,
                                     const float* __restrict__ hin,
                                     const float* __restrict__ cin,
                                     const float* __restrict__ bx,
                                     const unsigned short* __restrict__ Wp,
                                     float* __restrict__ out) {
    // sT: halo tile [3][66][72] bf16 (28512 B), sG: gate tile [128][68] f32 (34816 B), aliased
    __shared__ __align__(16) unsigned char smem[CH4 * GPAD * 4];
    unsigned short* sT = (unsigned short*)smem;
    float*          sG = (float*)smem;

    const int t    = threadIdx.x;
    const int wave = t >> 5;                 // 0..7 : M-tile
    const int lane = t & 31;

    const int tile = blockIdx.x & 1023;      // 1024 pixel-tiles per batch image
    const int b    = blockIdx.x >> 10;
    const int yrow = tile >> 2;              // 64 pixels share one image row
    const int x0   = (tile & 3) * TILE_N;

    // ================= stage raw halo tile (once) =================
    // zero-fill (covers channel pad + image-border padding)
    {
        uint4v z = {0u, 0u, 0u, 0u};
        uint4v* zp = (uint4v*)sT;
        #pragma unroll
        for (int i = t; i < (TROWS * TCOLS * TCH * 2) / 16; i += 256) zp[i] = z;
    }
    __syncthreads();

    // thread t < 198 owns one (row,col) halo position; loops channels with imm offsets
    if (t < TROWS * TCOLS) {
        int row = (t >= 132) ? 2 : (t >= 66 ? 1 : 0);
        int col = t - row * 66;
        int yy  = yrow - 1 + row;
        int xx  = x0 - 1 + col;
        if ((unsigned)yy < IMG && (unsigned)xx < IMG) {
            const float* xp = xin + ((size_t)b * CIN * IMG + yy) * IMG + xx;
            const float* hp = hin + ((size_t)b * HCC * IMG + yy) * IMG + xx;
            unsigned short* tp = sT + t * TCH;
            #pragma unroll
            for (int ch = 0; ch < CIN; ++ch) tp[ch]       = f2bf(xp[ch * PLANE]);
            #pragma unroll
            for (int ch = 0; ch < HCC; ++ch) tp[CIN + ch] = f2bf(hp[ch * PLANE]);
        }
    }
    __syncthreads();

    // ================= K loop: pure WMMA, no barriers =================
    const int mrow  = wave * 16 + (lane & 15);
    const int khalf = lane >> 4;                       // 0/1
    const unsigned short* aBase = Wp + (size_t)mrow * KTOT + khalf * 8;
    __builtin_prefetch(aBase, 0, 0);
    // one lane-dependent LDS base; everything else is immediate offsets
    const unsigned short* bLane = sT + (lane & 15) * TCH + khalf * 8;

    v8f acc[4] = {};
    #pragma unroll
    for (int kc = 0; kc < KCH; ++kc) {
        const int tap = kc >> 1;                       // compile-time after unroll
        const int r   = tap / 3;
        const int s   = tap - r * 3;
        const int chh = (kc & 1) * 32;                 // channel half of the 64-slot

        union { v16bf v; uint4v q[2]; } A, B;
        const uint4v* ap = (const uint4v*)(aBase + kc * 32);
        A.q[0] = ap[0];                                // K = kb..kb+7
        A.q[1] = ap[2];                                // K = kb+16..kb+23

        #pragma unroll
        for (int nt = 0; nt < 4; ++nt) {
            const uint4v* bp =
                (const uint4v*)(bLane + ((r * 66 + s) + nt * 16) * TCH + chh);
            B.q[0] = bp[0];                            // ch  .. +7
            B.q[1] = bp[2];                            // ch+16.. +23
            acc[nt] = __builtin_amdgcn_wmma_f32_16x16x32_bf16(
                false, A.v, false, B.v, (short)0, acc[nt], false, false);
        }
    }
    __syncthreads();   // done reading sT; safe to alias sG

    // ================= stage g-tile per C/D VGPR layout =================
    #pragma unroll
    for (int nt = 0; nt < 4; ++nt) {
        int nl    = nt * 16 + (lane & 15);
        int mbase = wave * 16 + khalf * 8;
        #pragma unroll
        for (int g = 0; g < 8; ++g)
            sG[(mbase + g) * GPAD + nl] = acc[nt][g];
    }
    __syncthreads();

    // ================= LSTM gates: 64 px x 32 hidden ch =================
    const size_t pbase = (size_t)yrow * IMG + x0;
    const size_t half  = (size_t)8 * HCC * PLANE;      // ch|cc split in d_out
    #pragma unroll
    for (int j = 0; j < 8; ++j) {
        int lin = t + j * 256;
        int hc  = lin >> 6;
        int n   = lin & 63;
        float gf = sG[(0  + hc) * GPAD + n] + bx[0  + hc];
        float gi = sG[(32 + hc) * GPAD + n] + bx[32 + hc];
        float gc = sG[(64 + hc) * GPAD + n] + bx[64 + hc];
        float go = sG[(96 + hc) * GPAD + n] + bx[96 + hc];
        size_t idx = ((size_t)b * HCC + hc) * PLANE + pbase + n;
        float cv  = cin[idx];
        float f   = 1.f / (1.f + __expf(-gf));
        float i   = 1.f / (1.f + __expf(-gi));
        float cg  = tanhf(gc);
        float o   = 1.f / (1.f + __expf(-go));
        float ccv = f * cv + i * cg;
        float chv = o * tanhf(ccv);
        out[idx]        = chv;   // ch
        out[idx + half] = ccv;   // cc
    }
}

extern "C" void kernel_launch(void* const* d_in, const int* in_sizes, int n_in,
                              void* d_out, int out_size, void* d_ws, size_t ws_size,
                              hipStream_t stream) {
    (void)in_sizes; (void)n_in; (void)out_size; (void)ws_size;
    const float* x  = (const float*)d_in[0];
    const float* h  = (const float*)d_in[1];
    const float* c  = (const float*)d_in[2];
    const float* wx = (const float*)d_in[3];
    const float* bx = (const float*)d_in[4];
    const float* wh = (const float*)d_in[5];
    unsigned short* Wp = (unsigned short*)d_ws;
    float* out = (float*)d_out;

    pack_weights_kernel<<<(CH4 * KTOT + 255) / 256, 256, 0, stream>>>(wx, wh, Wp);
    convlstm_wmma_kernel<<<8 * 1024, 256, 0, stream>>>(x, h, c, bx, Wp, out);
}
